// GroupedFeatureEncoder_20031727468925
// MI455X (gfx1250) — matrix operand order repaired and checked by reading, not compiled, not run on hardware
//
#include <hip/hip_runtime.h>
#include <hip/hip_bf16.h>

typedef __bf16 bf16_t;
typedef __attribute__((ext_vector_type(16))) __bf16 v16bf;
typedef __attribute__((ext_vector_type(8)))  __bf16 v8bf;
typedef __attribute__((ext_vector_type(8)))  float  v8f;

__device__ __forceinline__ bf16_t f2bf(float f) {
  union { float f; unsigned u; } a; a.f = f;
  unsigned u = a.u;
  u += 0x7FFFu + ((u >> 16) & 1u);           // round-to-nearest-even
  union { unsigned short s; bf16_t b; } c;
  c.s = (unsigned short)(u >> 16);
  return c.b;
}

// A fragment: lane m = lane%16, half = lane/16.
// elements 0..7  <- A[m][kbase + 8*half .. +7]
// elements 8..15 <- A[m][kbase + 16 + 8*half .. +7]
__device__ __forceinline__ v16bf load_afrag(const bf16_t* aRow, int kbase, int half) {
  v8bf lo = *(const v8bf*)(aRow + kbase + 8 * half);
  v8bf hi = *(const v8bf*)(aRow + kbase + 16 + 8 * half);
  v16bf f;
#pragma unroll
  for (int i = 0; i < 8; ++i) { f[i] = lo[i]; f[i + 8] = hi[i]; }
  return f;
}

// B fragment: lane n = lane%16, half = lane/16.
// elements 0..7  <- W[n][kbase + 8*half .. +7]
// elements 8..15 <- W[n][kbase + 16 + 8*half .. +7]
__device__ __forceinline__ v16bf load_bfrag(const bf16_t* bRow, int kbase, int half) {
  v8bf lo = *(const v8bf*)(bRow + kbase + 8 * half);
  v8bf hi = *(const v8bf*)(bRow + kbase + 16 + 8 * half);
  v16bf f;
#pragma unroll
  for (int i = 0; i < 8; ++i) { f[i] = lo[i]; f[i + 8] = hi[i]; }
  return f;
}

__device__ __forceinline__ v8f wmma_bf16(v16bf a, v16bf b, v8f c) {
  return __builtin_amdgcn_wmma_f32_16x16x32_bf16(false, a, false, b, (short)0, c,
                                                 false, false);
}

__global__ void cast_bf16_kernel(const float* __restrict__ src,
                                 bf16_t* __restrict__ dst, int n) {
  int i = blockIdx.x * blockDim.x + threadIdx.x;
  if (i < n) dst[i] = f2bf(src[i]);
}

// ---------------- Encoder: out = relu(LN(A @ W^T + b)) ----------------
// One workgroup = 16 tokens x 128 output channels (8 waves x 16 cols).
template <bool OUT_BF16>
__global__ void enc_kernel(const float* __restrict__ x, int xOffBase, int xGroupStride, int K,
                           const bf16_t* __restrict__ W,
                           const float* __restrict__ bias,
                           const float* __restrict__ gain,
                           const float* __restrict__ beta,
                           void* __restrict__ outp, int outRowStride, int outColOff,
                           int outGroupStride) {
  __shared__ bf16_t ldsA[16 * (512 + 8)];
  __shared__ float res[16][128];
  __shared__ float ps[16][17], ps2[16][17];
  __shared__ float mu[16], rs[16];

  const int g    = blockIdx.y;
  const int tile = blockIdx.x;
  const int tid  = threadIdx.x;
  const int lane = tid & 31, wave = tid >> 5;
  const int Kp   = K + 8;

  const float*  xrow  = x + (size_t)tile * 16 * 3328 + xOffBase + g * xGroupStride;
  const bf16_t* Wg    = W + (size_t)g * 128 * K;
  const float*  bg    = bias + g * 128;
  const float*  gg    = gain + g * 128;
  const float*  betag = beta + g * 128;

  for (int idx = tid; idx < 16 * K; idx += 256) {
    int r = idx / K, c = idx - r * K;
    ldsA[r * Kp + c] = f2bf(xrow[(size_t)r * 3328 + c]);
  }
  __syncthreads();

  const int n = lane & 15, half = lane >> 4;
  const int col = wave * 16 + n;
  const bf16_t* aRow = &ldsA[n * Kp];
  const bf16_t* bRow = Wg + (size_t)col * K;

  v8f acc = {};
  for (int kb = 0; kb < K; kb += 32) {
    v16bf bf = load_bfrag(bRow, kb, half);
    v16bf af = load_afrag(aRow, kb, half);
    acc = wmma_bf16(af, bf, acc);
  }

  float b = bg[col];
#pragma unroll
  for (int i = 0; i < 8; ++i) res[i + 8 * half][col] = acc[i] + b;
  __syncthreads();

  const int row = tid >> 4, tc = tid & 15;
  float s = 0.f, s2 = 0.f;
#pragma unroll
  for (int i = 0; i < 8; ++i) {
    float v = res[row][tc * 8 + i];
    s += v; s2 += v * v;
  }
  ps[row][tc] = s; ps2[row][tc] = s2;
  __syncthreads();
  if (tid < 16) {
    float a = 0.f, bb2 = 0.f;
    for (int i = 0; i < 16; ++i) { a += ps[tid][i]; bb2 += ps2[tid][i]; }
    float m = a * (1.0f / 128.0f);
    float v = bb2 * (1.0f / 128.0f) - m * m;
    mu[tid] = m;
    rs[tid] = rsqrtf(v + 1e-5f);
  }
  __syncthreads();

  const float m_ = mu[row], r_ = rs[row];
  const size_t obase = (size_t)(tile * 16 + row) * outRowStride + outColOff +
                       (size_t)g * outGroupStride;
#pragma unroll
  for (int i = 0; i < 8; ++i) {
    int c = tc * 8 + i;
    float y = (res[row][c] - m_) * r_ * gg[c] + betag[c];
    y = fmaxf(y, 0.0f);
    if (OUT_BF16) ((bf16_t*)outp)[obase + c] = f2bf(y);
    else          ((float*)outp)[obase + c]  = y;
  }
}

// -------- QKV GEMM: (N*6,128) @ Wqkv^T(384,128) + b -> fp32; M-tile 64 --------
__global__ void qkv_kernel(const float* __restrict__ A, const bf16_t* __restrict__ W,
                           const float* __restrict__ bias, float* __restrict__ out) {
  const int K = 128, Kp = K + 8;
  __shared__ bf16_t ldsA[64 * (128 + 8)];
  const int tile = blockIdx.x, nb = blockIdx.y;
  const int tid = threadIdx.x, lane = tid & 31, wave = tid >> 5;

  const float* arow = A + (size_t)tile * 64 * 128;
  for (int idx = tid; idx < 64 * 128; idx += 256) {
    int r = idx >> 7, c = idx & 127;
    ldsA[r * Kp + c] = f2bf(arow[idx]);
  }
  __syncthreads();

  const int n = lane & 15, half = lane >> 4;
  const int col = nb * 128 + wave * 16 + n;
  const bf16_t* bRow = W + (size_t)col * K;

  v8f acc[4] = {{}, {}, {}, {}};
#pragma unroll
  for (int kb = 0; kb < K; kb += 32) {
    v16bf bf = load_bfrag(bRow, kb, half);
#pragma unroll
    for (int s = 0; s < 4; ++s) {
      v16bf af = load_afrag(&ldsA[(s * 16 + n) * Kp], kb, half);
      acc[s] = wmma_bf16(af, bf, acc[s]);
    }
  }

  float b = bias[col];
#pragma unroll
  for (int s = 0; s < 4; ++s)
#pragma unroll
    for (int i = 0; i < 8; ++i)
      out[(size_t)(tile * 64 + s * 16 + i + 8 * half) * 384 + col] = acc[s][i] + b;
}

// ---------------- Per-node attention: 2 heads, 6x6 softmax ----------------
__global__ void attn_kernel(const float* __restrict__ qkv, float* __restrict__ o) {
  __shared__ float qs[6 * 384];
  __shared__ float sc[2][6][6];
  const int node = blockIdx.x, tid = threadIdx.x;
  const float* base = qkv + (size_t)node * 2304;
  for (int i = tid; i < 2304; i += 128) qs[i] = base[i];
  __syncthreads();

  if (tid < 72) {
    int h = tid / 36, r = tid % 36, i = r / 6, j = r % 6;
    const float* qp = &qs[i * 384 + h * 64];
    const float* kp = &qs[j * 384 + 128 + h * 64];
    float s = 0.f;
    for (int d = 0; d < 64; ++d) s += qp[d] * kp[d];
    sc[h][i][j] = s * 0.125f;
  }
  __syncthreads();
  if (tid < 12) {
    int h = tid / 6, i = tid % 6;
    float mx = sc[h][i][0];
    for (int j = 1; j < 6; ++j) mx = fmaxf(mx, sc[h][i][j]);
    float e[6], sum = 0.f;
    for (int j = 0; j < 6; ++j) { e[j] = __expf(sc[h][i][j] - mx); sum += e[j]; }
    float inv = 1.0f / sum;
    for (int j = 0; j < 6; ++j) sc[h][i][j] = e[j] * inv;
  }
  __syncthreads();
  {
    int h = tid >> 6, d = tid & 63;
    for (int i = 0; i < 6; ++i) {
      float acc = 0.f;
      for (int j = 0; j < 6; ++j) acc += sc[h][i][j] * qs[j * 384 + 256 + h * 64 + d];
      o[(size_t)node * 768 + i * 128 + h * 64 + d] = acc;
    }
  }
}

// ------- attn proj + bias + residual + LN -> bf16 into concat[token*1792 + grp*128] -------
__global__ void proj_ln_kernel(const float* __restrict__ Ao, const bf16_t* __restrict__ W,
                               const float* __restrict__ bias,
                               const float* __restrict__ resid,
                               const float* __restrict__ gain,
                               const float* __restrict__ beta,
                               bf16_t* __restrict__ concat) {
  const int K = 128, Kp = K + 8;
  __shared__ bf16_t ldsA[16 * (128 + 8)];
  __shared__ float res[16][128];
  __shared__ float ps[16][17], ps2[16][17];
  __shared__ float mu[16], rs[16];

  const int tile = blockIdx.x, tid = threadIdx.x;
  const int lane = tid & 31, wave = tid >> 5;

  const float* arow = Ao + (size_t)tile * 16 * 128;
  for (int idx = tid; idx < 16 * 128; idx += 256) {
    int r = idx >> 7, c = idx & 127;
    ldsA[r * Kp + c] = f2bf(arow[idx]);
  }
  __syncthreads();

  const int n = lane & 15, half = lane >> 4;
  const int col = wave * 16 + n;
  const bf16_t* aRow = &ldsA[n * Kp];
  const bf16_t* bRow = W + (size_t)col * K;

  v8f acc = {};
#pragma unroll
  for (int kb = 0; kb < K; kb += 32) {
    v16bf bf = load_bfrag(bRow, kb, half);
    v16bf af = load_afrag(aRow, kb, half);
    acc = wmma_bf16(af, bf, acc);
  }

  float b = bias[col];
#pragma unroll
  for (int i = 0; i < 8; ++i) {
    int row = i + 8 * half;
    res[row][col] = acc[i] + b + resid[(size_t)(tile * 16 + row) * 128 + col];
  }
  __syncthreads();

  const int row = tid >> 4, tc = tid & 15;
  float s = 0.f, s2 = 0.f;
#pragma unroll
  for (int i = 0; i < 8; ++i) {
    float v = res[row][tc * 8 + i];
    s += v; s2 += v * v;
  }
  ps[row][tc] = s; ps2[row][tc] = s2;
  __syncthreads();
  if (tid < 16) {
    float a = 0.f, bb2 = 0.f;
    for (int i = 0; i < 16; ++i) { a += ps[tid][i]; bb2 += ps2[tid][i]; }
    float m = a * (1.0f / 128.0f);
    float v = bb2 * (1.0f / 128.0f) - m * m;
    mu[tid] = m;
    rs[tid] = rsqrtf(v + 1e-5f);
  }
  __syncthreads();

  const float m_ = mu[row], r_ = rs[row];
  const int rg = tile * 16 + row;
  const int token = rg / 6, grp = rg % 6;
  bf16_t* crow = concat + (size_t)token * 1792 + grp * 128;
#pragma unroll
  for (int i = 0; i < 8; ++i) {
    int c = tc * 8 + i;
    crow[c] = f2bf((res[row][c] - m_) * r_ * gain[c] + beta[c]);
  }
}

// ---- Final GEMM: concat(16384x1792) @ Wa^T(1024x1792) + ba -> fp32; M-tile 64 ----
__global__ void final_kernel(const bf16_t* __restrict__ A, const bf16_t* __restrict__ W,
                             const float* __restrict__ bias, float* __restrict__ out) {
  const int K = 1792, KC = 256, Kp = KC + 8;
  __shared__ bf16_t ldsA[64 * (256 + 8)];
  const int tile = blockIdx.x, nb = blockIdx.y;
  const int tid = threadIdx.x, lane = tid & 31, wave = tid >> 5;
  const int n = lane & 15, half = lane >> 4;
  const int col = nb * 128 + wave * 16 + n;
  const bf16_t* bRowBase = W + (size_t)col * K;

  v8f acc[4] = {{}, {}, {}, {}};
  for (int kc = 0; kc < K; kc += KC) {
    const bf16_t* abase = A + (size_t)tile * 64 * K + kc;
    for (int idx = tid; idx < 64 * (KC / 2); idx += 256) {
      int r = idx >> 7, c2 = idx & 127;
      ((uint32_t*)(&ldsA[r * Kp]))[c2] = ((const uint32_t*)(abase + (size_t)r * K))[c2];
    }
    __syncthreads();
    for (int kb = 0; kb < KC; kb += 32) {
      v16bf bf = load_bfrag(bRowBase + kc, kb, half);
#pragma unroll
      for (int s = 0; s < 4; ++s) {
        v16bf af = load_afrag(&ldsA[(s * 16 + n) * Kp], kb, half);
        acc[s] = wmma_bf16(af, bf, acc[s]);
      }
    }
    __syncthreads();
  }
  float b = bias[col];
#pragma unroll
  for (int s = 0; s < 4; ++s)
#pragma unroll
    for (int i = 0; i < 8; ++i)
      out[(size_t)(tile * 64 + s * 16 + i + 8 * half) * 1024 + col] = acc[s][i] + b;
}

extern "C" void kernel_launch(void* const* d_in, const int* in_sizes, int n_in,
                              void* d_out, int out_size, void* d_ws, size_t ws_size,
                              hipStream_t stream) {
  (void)in_sizes; (void)n_in; (void)out_size; (void)ws_size;
  const float* x     = (const float*)d_in[0];
  const float* Wp    = (const float*)d_in[1];
  const float* bp    = (const float*)d_in[2];
  const float* gp    = (const float*)d_in[3];
  const float* betap = (const float*)d_in[4];
  const float* Wo    = (const float*)d_in[5];
  const float* bo    = (const float*)d_in[6];
  const float* go    = (const float*)d_in[7];
  const float* betao = (const float*)d_in[8];
  const float* Wb    = (const float*)d_in[9];
  const float* bb    = (const float*)d_in[10];
  const float* gb    = (const float*)d_in[11];
  const float* betab = (const float*)d_in[12];
  const float* We    = (const float*)d_in[13];
  const float* be    = (const float*)d_in[14];
  const float* ge    = (const float*)d_in[15];
  const float* betae = (const float*)d_in[16];
  const float* Wqkv  = (const float*)d_in[17];
  const float* bqkv  = (const float*)d_in[18];
  const float* Wattn = (const float*)d_in[19];
  const float* battn = (const float*)d_in[20];
  const float* gn    = (const float*)d_in[21];
  const float* betan = (const float*)d_in[22];
  const float* Wa    = (const float*)d_in[23];
  const float* ba    = (const float*)d_in[24];
  float* out = (float*)d_out;

  char* ws = (char*)d_ws;
  auto alloc = [&](size_t bytes) -> char* {
    char* p = ws;
    ws += (bytes + 255) & ~(size_t)255;
    return p;
  };
  float*  hp_f32  = (float*)alloc((size_t)16384 * 768 * 4);
  float*  qkv_f32 = (float*)alloc((size_t)98304 * 384 * 4);
  float*  o_f32   = (float*)alloc((size_t)16384 * 768 * 4);
  bf16_t* concat  = (bf16_t*)alloc((size_t)16384 * 1792 * 2);
  bf16_t* Wp_b  = (bf16_t*)alloc((size_t)6 * 128 * 256 * 2);
  bf16_t* Wo_b  = (bf16_t*)alloc((size_t)6 * 128 * 192 * 2);
  bf16_t* Wb_b  = (bf16_t*)alloc((size_t)128 * 512 * 2);
  bf16_t* We_b  = (bf16_t*)alloc((size_t)128 * 128 * 2);
  bf16_t* Wq_b  = (bf16_t*)alloc((size_t)384 * 128 * 2);
  bf16_t* Wat_b = (bf16_t*)alloc((size_t)128 * 128 * 2);
  bf16_t* Wa_b  = (bf16_t*)alloc((size_t)1024 * 1792 * 2);

  auto cast = [&](const float* s, bf16_t* d, int nelem) {
    cast_bf16_kernel<<<(nelem + 255) / 256, 256, 0, stream>>>(s, d, nelem);
  };
  cast(Wp,    Wp_b,  6 * 128 * 256);
  cast(Wo,    Wo_b,  6 * 128 * 192);
  cast(Wb,    Wb_b,  128 * 512);
  cast(We,    We_b,  128 * 128);
  cast(Wqkv,  Wq_b,  384 * 128);
  cast(Wattn, Wat_b, 128 * 128);
  cast(Wa,    Wa_b,  1024 * 1792);

  // Encoders: 1024 token-tiles of 16
  enc_kernel<false><<<dim3(1024, 6), 256, 0, stream>>>(
      x, 0, 256, 256, Wp_b, bp, gp, betap, hp_f32, 768, 0, 128);
  enc_kernel<true><<<dim3(1024, 6), 256, 0, stream>>>(
      x, 1536, 192, 192, Wo_b, bo, go, betao, concat, 1792, 768, 128);
  enc_kernel<true><<<dim3(1024, 1), 256, 0, stream>>>(
      x, 2688, 0, 512, Wb_b, bb, gb, betab, concat, 1792, 1536, 0);
  enc_kernel<true><<<dim3(1024, 1), 256, 0, stream>>>(
      x, 3200, 0, 128, We_b, be, ge, betae, concat, 1792, 1664, 0);

  qkv_kernel<<<dim3(1536, 3), 256, 0, stream>>>(hp_f32, Wq_b, bqkv, qkv_f32);
  attn_kernel<<<16384, 128, 0, stream>>>(qkv_f32, o_f32);
  proj_ln_kernel<<<6144, 256, 0, stream>>>(o_f32, Wat_b, battn, hp_f32, gn, betan, concat);
  final_kernel<<<dim3(256, 8), 256, 0, stream>>>(concat, Wa_b, ba, out);
}
